// MLP_72645076844940
// MI455X (gfx1250) — matrix operand order repaired and checked by reading, not compiled, not run on hardware
//
#include <hip/hip_runtime.h>

#define NROWS  16384
#define DIN    360
#define HDIM   512
#define MSLOTS 2048
#define KSTEP  64

typedef __bf16 bf16_t;
typedef __attribute__((ext_vector_type(16))) __bf16 v16bf;
typedef __attribute__((ext_vector_type(8)))  __bf16 v8bf;
typedef __attribute__((ext_vector_type(4)))  __bf16 v4bf;
typedef __attribute__((ext_vector_type(8)))  float  v8f;

__device__ inline v8f vzero8() {
  v8f z;
#pragma unroll
  for (int i = 0; i < 8; ++i) z[i] = 0.0f;
  return z;
}

// A-fragment (16-bit A 16x32, ISA 7.12.2): lane half selects K groups
// {8h..8h+7} and {16+8h..23+8h} -> two contiguous 16B LDS reads.
__device__ inline v16bf frag_a(const bf16_t* p, int half) {
  const v8bf lo = *(const v8bf*)(p + half * 8);
  const v8bf hi = *(const v8bf*)(p + 16 + half * 8);
  return __builtin_shufflevector(lo, hi, 0, 1, 2, 3, 4, 5, 6, 7,
                                 8, 9, 10, 11, 12, 13, 14, 15);
}

// B-fragment (16-bit B 32x16): lane half selects K groups
// {8h..8h+7} and {16+8h..23+8h}, matching frag_a.
__device__ inline v16bf frag_b(const bf16_t* p, int half) {
  const v8bf lo = *(const v8bf*)(p + half * 8);
  const v8bf hi = *(const v8bf*)(p + 16 + half * 8);
  return __builtin_shufflevector(lo, hi, 0, 1, 2, 3, 4, 5, 6, 7,
                                 8, 9, 10, 11, 12, 13, 14, 15);
}

// Async global->LDS copy, 16 bytes per lane (ISA 15.18.3 op 98, ASYNCcnt).
// LDS operand = low 32 bits of the flat shared-memory address (ISA 10.2).
__device__ inline void async_copy_b128(void* lds, const void* gaddr) {
  unsigned lo = (unsigned)(unsigned long long)lds;
  asm volatile("global_load_async_to_lds_b128 %0, %1, off"
               :: "v"(lo), "v"(gaddr)
               : "memory");
}
__device__ inline void wait_async0() {
  asm volatile("s_wait_asynccnt 0x0" ::: "memory");
}

// ---------------------------------------------------------------------------
// C[n,ho] = act(A[n,k] @ W[ho,k]^T + bias)
// Block tile 128x64, 8 waves (4x2), wave tile 32x32, K-step 64 (2 WMMA slabs).
// fp32 -> bf16 conversion fused into float4 staging; f32 accumulate.
// ---------------------------------------------------------------------------
template<bool BIAS, bool RELU>
__global__ __launch_bounds__(256)
void gemm_nt_wmma(const float* __restrict__ A, const float* __restrict__ W,
                  const float* __restrict__ bias, float* __restrict__ C,
                  int n, int k, int ho)
{
  __shared__ __align__(16) bf16_t As[128 * KSTEP];   // 16 KB
  __shared__ __align__(16) bf16_t Bs[64 * KSTEP];    //  8 KB

  const int tid  = threadIdx.x;
  const int lane = tid & 31;
  const int wave = tid >> 5;
  const int wr   = wave & 3;
  const int wc   = wave >> 2;
  const int half = lane >> 4;
  const int l16  = lane & 15;

  const int rowBlk = blockIdx.x * 128;
  const int colBlk = blockIdx.y * 64;

  v8f acc[2][2];
#pragma unroll
  for (int i = 0; i < 2; ++i)
#pragma unroll
    for (int j = 0; j < 2; ++j) acc[i][j] = vzero8();

  for (int k0 = 0; k0 < k; k0 += KSTEP) {
    if (k0 + KSTEP < k) {
      __builtin_prefetch(&A[(size_t)(rowBlk + (tid & 127)) * k + k0 + KSTEP], 0, 0);
      if (tid < 64)
        __builtin_prefetch(&W[(size_t)(colBlk + tid) * k + k0 + KSTEP], 0, 0);
    }

    // Stage A tile 128x64: float4 global loads -> packed v4bf LDS stores.
#pragma unroll
    for (int e = 0; e < 8; ++e) {
      int vid = tid + e * 256;          // 0..2047
      int r   = vid >> 4;               // 0..127
      int kk  = (vid & 15) << 2;        // 0..60
      int gk  = k0 + kk;
      v4bf p;
      if (gk + 3 < k) {
        const float4 f = *(const float4*)&A[(size_t)(rowBlk + r) * k + gk];
        p[0] = (bf16_t)f.x; p[1] = (bf16_t)f.y;
        p[2] = (bf16_t)f.z; p[3] = (bf16_t)f.w;
      } else {
#pragma unroll
        for (int t = 0; t < 4; ++t) {
          float v = (gk + t < k) ? A[(size_t)(rowBlk + r) * k + gk + t] : 0.0f;
          p[t] = (bf16_t)v;
        }
      }
      *(v4bf*)&As[r * KSTEP + kk] = p;
    }
    // Stage B tile 64x64: Bs[c][kk] = W[colBlk+c][k0+kk]
#pragma unroll
    for (int e = 0; e < 4; ++e) {
      int vid = tid + e * 256;          // 0..1023
      int c   = vid >> 4;               // 0..63
      int kk  = (vid & 15) << 2;        // 0..60
      int gk  = k0 + kk;
      v4bf p;
      if (gk + 3 < k) {
        const float4 f = *(const float4*)&W[(size_t)(colBlk + c) * k + gk];
        p[0] = (bf16_t)f.x; p[1] = (bf16_t)f.y;
        p[2] = (bf16_t)f.z; p[3] = (bf16_t)f.w;
      } else {
#pragma unroll
        for (int t = 0; t < 4; ++t) {
          float v = (gk + t < k) ? W[(size_t)(colBlk + c) * k + gk + t] : 0.0f;
          p[t] = (bf16_t)v;
        }
      }
      *(v4bf*)&Bs[c * KSTEP + kk] = p;
    }
    __syncthreads();

#pragma unroll
    for (int ks = 0; ks < KSTEP; ks += 32) {
      v16bf af[2], bfr[2];
#pragma unroll
      for (int i = 0; i < 2; ++i)
        af[i] = frag_a(&As[(wr * 32 + i * 16 + l16) * KSTEP + ks], half);
#pragma unroll
      for (int j = 0; j < 2; ++j)
        bfr[j] = frag_b(&Bs[(wc * 32 + j * 16 + l16) * KSTEP + ks], half);
#pragma unroll
      for (int i = 0; i < 2; ++i)
#pragma unroll
        for (int j = 0; j < 2; ++j)
          acc[i][j] = __builtin_amdgcn_wmma_f32_16x16x32_bf16(
              false, af[i], false, bfr[j], (short)0, acc[i][j], false, false);
    }
    __syncthreads();
  }

#pragma unroll
  for (int i = 0; i < 2; ++i)
#pragma unroll
    for (int j = 0; j < 2; ++j)
#pragma unroll
      for (int v = 0; v < 8; ++v) {
        int row = rowBlk + wr * 32 + i * 16 + v + 8 * half;
        int col = colBlk + wc * 32 + j * 16 + l16;
        float x = acc[i][j][v];
        if (BIAS) x += bias[col];
        if (RELU) x = fmaxf(x, 0.0f);
        C[(size_t)row * ho + col] = x;
      }
}

// ---------------------------------------------------------------------------
// S = Ab @ Bb^T with bf16 operands resident in global memory.
// Staging is a raw byte copy -> use global_load_async_to_lds_b128 (ASYNCcnt).
// ---------------------------------------------------------------------------
__global__ __launch_bounds__(256)
void scores_gemm_async_wmma(const bf16_t* __restrict__ Ab,   // [NROWS, HDIM]
                            const bf16_t* __restrict__ Bb,   // [MSLOTS, HDIM]
                            float* __restrict__ S)           // [NROWS, MSLOTS]
{
  __shared__ __align__(16) bf16_t As[128 * KSTEP];   // 16 KB
  __shared__ __align__(16) bf16_t Bs[64 * KSTEP];    //  8 KB

  const int tid  = threadIdx.x;
  const int lane = tid & 31;
  const int wave = tid >> 5;
  const int wr   = wave & 3;
  const int wc   = wave >> 2;
  const int half = lane >> 4;
  const int l16  = lane & 15;

  const int rowBlk = blockIdx.x * 128;
  const int colBlk = blockIdx.y * 64;

  v8f acc[2][2];
#pragma unroll
  for (int i = 0; i < 2; ++i)
#pragma unroll
    for (int j = 0; j < 2; ++j) acc[i][j] = vzero8();

  for (int k0 = 0; k0 < HDIM; k0 += KSTEP) {
    // A tile: 128 rows x 128 B = 1024 x 16B segments, 4 per thread.
#pragma unroll
    for (int e = 0; e < 4; ++e) {
      int seg = tid + e * 256;           // 0..1023
      int r   = seg >> 3;                // 0..127
      int s8  = (seg & 7) << 3;          // 0,8,...,56 (bf16 elems)
      async_copy_b128(&As[r * KSTEP + s8],
                      &Ab[(size_t)(rowBlk + r) * HDIM + k0 + s8]);
    }
    // B tile: 64 rows x 128 B = 512 x 16B segments, 2 per thread.
#pragma unroll
    for (int e = 0; e < 2; ++e) {
      int seg = tid + e * 256;           // 0..511
      int c   = seg >> 3;                // 0..63
      int s8  = (seg & 7) << 3;
      async_copy_b128(&Bs[c * KSTEP + s8],
                      &Bb[(size_t)(colBlk + c) * HDIM + k0 + s8]);
    }
    wait_async0();        // each wave drains its own ASYNCcnt...
    __syncthreads();      // ...then the barrier makes all waves' data visible

#pragma unroll
    for (int ks = 0; ks < KSTEP; ks += 32) {
      v16bf af[2], bfr[2];
#pragma unroll
      for (int i = 0; i < 2; ++i)
        af[i] = frag_a(&As[(wr * 32 + i * 16 + l16) * KSTEP + ks], half);
#pragma unroll
      for (int j = 0; j < 2; ++j)
        bfr[j] = frag_b(&Bs[(wc * 32 + j * 16 + l16) * KSTEP + ks], half);
#pragma unroll
      for (int i = 0; i < 2; ++i)
#pragma unroll
        for (int j = 0; j < 2; ++j)
          acc[i][j] = __builtin_amdgcn_wmma_f32_16x16x32_bf16(
              false, af[i], false, bfr[j], (short)0, acc[i][j], false, false);
    }
    __syncthreads();
  }

#pragma unroll
  for (int i = 0; i < 2; ++i)
#pragma unroll
    for (int j = 0; j < 2; ++j)
#pragma unroll
      for (int v = 0; v < 8; ++v) {
        int row = rowBlk + wr * 32 + i * 16 + v + 8 * half;
        int col = colBlk + wc * 32 + j * 16 + l16;
        S[(size_t)row * MSLOTS + col] = acc[i][j][v];
      }
}

// fp32 -> bf16 bulk conversion (n multiple of 4)
__global__ __launch_bounds__(256)
void cvt_f32_bf16_kernel(const float* __restrict__ src, bf16_t* __restrict__ dst, int n)
{
  int base = (blockIdx.x * blockDim.x + threadIdx.x) * 4;
  if (base + 3 < n) {
    const float4 f = *(const float4*)&src[base];
    v4bf p;
    p[0] = (bf16_t)f.x; p[1] = (bf16_t)f.y;
    p[2] = (bf16_t)f.z; p[3] = (bf16_t)f.w;
    *(v4bf*)&dst[base] = p;
  }
}

// ---------------------------------------------------------------------------
// Per-row stats of S: max (+first argmax), -min, sum exp(S-max), sum exp(-S+min)
// ---------------------------------------------------------------------------
__global__ __launch_bounds__(256)
void row_stats_kernel(const float* __restrict__ S,
                      float* __restrict__ rmax, float* __restrict__ rsumP,
                      float* __restrict__ rnmax, float* __restrict__ rsumN,
                      int* __restrict__ g)
{
  const int i = blockIdx.x;
  const float* row = S + (size_t)i * MSLOTS;
  __shared__ float smax[256];
  __shared__ float smin[256];
  __shared__ int   sarg[256];
  __shared__ float ssum[256];

  const int tid = threadIdx.x;
  float mx = -3.4e38f, mn = 3.4e38f;
  int am = 0x7fffffff;
  for (int j = tid; j < MSLOTS; j += 256) {
    float v = row[j];
    if (v > mx) { mx = v; am = j; }
    mn = fminf(mn, v);
  }
  smax[tid] = mx; smin[tid] = mn; sarg[tid] = am;
  __syncthreads();
  for (int s = 128; s > 0; s >>= 1) {
    if (tid < s) {
      float om = smax[tid + s]; int oa = sarg[tid + s];
      if (om > smax[tid] || (om == smax[tid] && oa < sarg[tid])) {
        smax[tid] = om; sarg[tid] = oa;
      }
      smin[tid] = fminf(smin[tid], smin[tid + s]);
    }
    __syncthreads();
  }
  const float MX  = smax[0];
  const float NMX = -smin[0];
  const int   AM  = sarg[0];

  float sp = 0.f, sn = 0.f;
  for (int j = tid; j < MSLOTS; j += 256) {
    float v = row[j];
    sp += __expf(v - MX);
    sn += __expf(-v - NMX);
  }
  ssum[tid] = sp; __syncthreads();
  for (int s = 128; s > 0; s >>= 1) { if (tid < s) ssum[tid] += ssum[tid + s]; __syncthreads(); }
  const float SP = ssum[0];
  __syncthreads();
  ssum[tid] = sn; __syncthreads();
  for (int s = 128; s > 0; s >>= 1) { if (tid < s) ssum[tid] += ssum[tid + s]; __syncthreads(); }
  if (tid == 0) {
    rmax[i] = MX; rnmax[i] = NMX; rsumP[i] = SP; rsumN[i] = ssum[0]; g[i] = AM;
  }
}

// ---------------------------------------------------------------------------
// Column max of S. col_max(softmax(S,axis=0))[j] == 1/csum_j, only cmax needed.
// ---------------------------------------------------------------------------
__global__ __launch_bounds__(256)
void col_max_kernel(const float* __restrict__ S, float* __restrict__ cmax)
{
  const int col = blockIdx.x * 64 + (threadIdx.x & 63);
  const int rg  = threadIdx.x >> 6;   // 0..3
  float mx = -3.4e38f;
  for (int i = rg; i < NROWS; i += 4)
    mx = fmaxf(mx, S[(size_t)i * MSLOTS + col]);
  __shared__ float sm[256];
  sm[threadIdx.x] = mx;
  __syncthreads();
  if (rg == 0)
    cmax[col] = fmaxf(fmaxf(sm[threadIdx.x], sm[threadIdx.x + 64]),
                      fmaxf(sm[threadIdx.x + 128], sm[threadIdx.x + 192]));
}

__global__ void zero_kernel(float* __restrict__ p, int n) {
  int i = blockIdx.x * blockDim.x + threadIdx.x;
  if (i < n) p[i] = 0.0f;
}

// w_i = exp(rmax_i - cmax[g_i]); qupd[g_i,:] += w_i * h[i,:]
__global__ __launch_bounds__(256)
void upload_scatter_kernel(const float* __restrict__ h, const float* __restrict__ rmax,
                           const float* __restrict__ cmax, const int* __restrict__ g,
                           float* __restrict__ qupd)
{
  const int i = blockIdx.x;
  const int j = g[i];
  const float w = __expf(rmax[i] - cmax[j]);
  for (int c = threadIdx.x; c < HDIM; c += 256)
    atomicAdd(&qupd[(size_t)j * HDIM + c], w * h[(size_t)i * HDIM + c]);
}

// mem[j,:] = normalize(qupd[j,:] + Kmem[j,:])  (passthrough if train==0)
__global__ __launch_bounds__(256)
void mem_finalize_kernel(const float* __restrict__ qupd, const float* __restrict__ Kmem,
                         const int* __restrict__ train, float* __restrict__ outMem)
{
  const int j = blockIdx.x;
  if (*train == 0) {
    for (int c = threadIdx.x; c < HDIM; c += 256)
      outMem[(size_t)j * HDIM + c] = Kmem[(size_t)j * HDIM + c];
    return;
  }
  __shared__ float u[HDIM];
  __shared__ float red[256];
  float ss = 0.f;
  for (int c = threadIdx.x; c < HDIM; c += 256) {
    float v = qupd[(size_t)j * HDIM + c] + Kmem[(size_t)j * HDIM + c];
    u[c] = v; ss += v * v;
  }
  red[threadIdx.x] = ss; __syncthreads();
  for (int s = 128; s > 0; s >>= 1) {
    if (threadIdx.x < s) red[threadIdx.x] += red[threadIdx.x + s];
    __syncthreads();
  }
  const float scale = 1.0f / fmaxf(sqrtf(red[0]), 1e-12f);
  for (int c = threadIdx.x; c < HDIM; c += 256)
    outMem[(size_t)j * HDIM + c] = u[c] * scale;
}

// ---------------------------------------------------------------------------
// Fused: outP = h * (softmax(S,1) @ Kmem), outN = h * (softmax(-S,1) @ Kmem)
// A tiles built on-the-fly as bf16 exp(S - rmax) / exp(-S - rnmax); K-step 64.
// ---------------------------------------------------------------------------
__global__ __launch_bounds__(256)
void attn_pv_wmma(const float* __restrict__ S, const float* __restrict__ Kmem,
                  const float* __restrict__ h,
                  const float* __restrict__ rmax, const float* __restrict__ rsumP,
                  const float* __restrict__ rnmax, const float* __restrict__ rsumN,
                  float* __restrict__ outP, float* __restrict__ outN)
{
  __shared__ __align__(16) bf16_t Ap[128 * KSTEP];   // 16 KB
  __shared__ __align__(16) bf16_t An[128 * KSTEP];   // 16 KB
  __shared__ __align__(16) bf16_t Bs[64 * KSTEP];    //  8 KB
  __shared__ float rm[128], rn[128];

  const int tid  = threadIdx.x;
  const int lane = tid & 31;
  const int wave = tid >> 5;
  const int wr   = wave & 3;
  const int wc   = wave >> 2;
  const int half = lane >> 4;
  const int l16  = lane & 15;
  const int rowBlk = blockIdx.x * 128;
  const int colBlk = blockIdx.y * 64;

  if (tid < 128) { rm[tid] = rmax[rowBlk + tid]; rn[tid] = rnmax[rowBlk + tid]; }
  __syncthreads();

  v8f accP[2][2], accN[2][2];
#pragma unroll
  for (int i = 0; i < 2; ++i)
#pragma unroll
    for (int j = 0; j < 2; ++j) { accP[i][j] = vzero8(); accN[i][j] = vzero8(); }

  for (int k0 = 0; k0 < MSLOTS; k0 += KSTEP) {
    if (k0 + KSTEP < MSLOTS)
      __builtin_prefetch(&S[(size_t)(rowBlk + (tid & 127)) * MSLOTS + k0 + KSTEP], 0, 0);

    // Stage exp tiles from S (float4) -> bf16.
#pragma unroll
    for (int e = 0; e < 8; ++e) {
      int vid = tid + e * 256;          // 0..2047
      int r   = vid >> 4;               // 0..127
      int kk  = (vid & 15) << 2;        // 0..60
      const float4 f = *(const float4*)&S[(size_t)(rowBlk + r) * MSLOTS + k0 + kk];
      const float* fp = &f.x;
      v4bf pp, pn;
#pragma unroll
      for (int t = 0; t < 4; ++t) {
        pp[t] = (bf16_t)__expf(fp[t] - rm[r]);
        pn[t] = (bf16_t)__expf(-fp[t] - rn[r]);
      }
      *(v4bf*)&Ap[r * KSTEP + kk] = pp;
      *(v4bf*)&An[r * KSTEP + kk] = pn;
    }
    // Stage B tile (transpose of Kmem rows): Bs[c][kk] = Kmem[k0+kk][colBlk+c]
#pragma unroll
    for (int e = 0; e < 4; ++e) {
      int vid = tid + e * 256;          // 0..1023
      int kk  = vid >> 4;               // 0..63
      int c4  = (vid & 15) << 2;        // 0..60
      const float4 f = *(const float4*)&Kmem[(size_t)(k0 + kk) * HDIM + colBlk + c4];
      Bs[(c4 + 0) * KSTEP + kk] = (bf16_t)f.x;
      Bs[(c4 + 1) * KSTEP + kk] = (bf16_t)f.y;
      Bs[(c4 + 2) * KSTEP + kk] = (bf16_t)f.z;
      Bs[(c4 + 3) * KSTEP + kk] = (bf16_t)f.w;
    }
    __syncthreads();

#pragma unroll
    for (int ks = 0; ks < KSTEP; ks += 32) {
      v16bf ap[2], an[2], bfr[2];
#pragma unroll
      for (int i = 0; i < 2; ++i) {
        ap[i] = frag_a(&Ap[(wr * 32 + i * 16 + l16) * KSTEP + ks], half);
        an[i] = frag_a(&An[(wr * 32 + i * 16 + l16) * KSTEP + ks], half);
      }
#pragma unroll
      for (int j = 0; j < 2; ++j)
        bfr[j] = frag_b(&Bs[(wc * 32 + j * 16 + l16) * KSTEP + ks], half);
#pragma unroll
      for (int i = 0; i < 2; ++i)
#pragma unroll
        for (int j = 0; j < 2; ++j) {
          accP[i][j] = __builtin_amdgcn_wmma_f32_16x16x32_bf16(
              false, ap[i], false, bfr[j], (short)0, accP[i][j], false, false);
          accN[i][j] = __builtin_amdgcn_wmma_f32_16x16x32_bf16(
              false, an[i], false, bfr[j], (short)0, accN[i][j], false, false);
        }
    }
    __syncthreads();
  }

#pragma unroll
  for (int i = 0; i < 2; ++i)
#pragma unroll
    for (int j = 0; j < 2; ++j)
#pragma unroll
      for (int v = 0; v < 8; ++v) {
        int row = rowBlk + wr * 32 + i * 16 + v + 8 * half;
        int col = colBlk + wc * 32 + j * 16 + l16;
        size_t o = (size_t)row * HDIM + col;
        float gate = h[o];
        outP[o] = gate * accP[i][j][v] / rsumP[row];
        outN[o] = gate * accN[i][j][v] / rsumN[row];
      }
}

// pred[i] = dot(h[i,:], Wd) + bd  (one wave per row)
__global__ __launch_bounds__(256)
void pred_dot_kernel(const float* __restrict__ h, const float* __restrict__ Wd,
                     const float* __restrict__ bd, float* __restrict__ pred)
{
  const int row  = blockIdx.x * 8 + (threadIdx.x >> 5);
  const int lane = threadIdx.x & 31;
  float s = 0.f;
  for (int c = lane; c < HDIM; c += 32)
    s += h[(size_t)row * HDIM + c] * Wd[c];
#pragma unroll
  for (int off = 16; off > 0; off >>= 1) s += __shfl_down(s, off, 32);
  if (lane == 0) pred[row] = s + bd[0];
}

// ---------------------------------------------------------------------------
extern "C" void kernel_launch(void* const* d_in, const int* in_sizes, int n_in,
                              void* d_out, int out_size, void* d_ws, size_t ws_size,
                              hipStream_t stream)
{
  const float* x     = (const float*)d_in[0];
  const float* Kmem  = (const float*)d_in[1];
  const float* W1    = (const float*)d_in[2];
  const float* b1    = (const float*)d_in[3];
  const float* W2    = (const float*)d_in[4];
  const float* b2    = (const float*)d_in[5];
  const float* W3    = (const float*)d_in[6];
  const float* b3    = (const float*)d_in[7];
  const float* Wd    = (const float*)d_in[8];
  const float* bd    = (const float*)d_in[9];
  const int*   train = (const int*)d_in[10];

  float* out     = (float*)d_out;
  float* outPred = out;                                   // [N]
  float* outN    = out + NROWS;                           // [N,H]
  float* outP    = outN + (size_t)NROWS * HDIM;           // [N,H]
  float* outMem  = outP + (size_t)NROWS * HDIM;           // [M,H]

  char* w = (char*)d_ws;
  float*  S     = (float*)w;   w += (size_t)NROWS * MSLOTS * sizeof(float);
  float*  buf0  = (float*)w;   w += (size_t)NROWS * HDIM * sizeof(float);
  float*  buf1  = (float*)w;   w += (size_t)NROWS * HDIM * sizeof(float);
  float*  qupd  = (float*)w;   w += (size_t)MSLOTS * HDIM * sizeof(float);
  bf16_t* hb    = (bf16_t*)w;  w += (size_t)NROWS * HDIM * sizeof(bf16_t);
  bf16_t* kb    = (bf16_t*)w;  w += (size_t)MSLOTS * HDIM * sizeof(bf16_t);
  float*  rmax  = (float*)w;   w += NROWS * sizeof(float);
  float*  rsumP = (float*)w;   w += NROWS * sizeof(float);
  float*  rnmax = (float*)w;   w += NROWS * sizeof(float);
  float*  rsumN = (float*)w;   w += NROWS * sizeof(float);
  float*  cmax  = (float*)w;   w += MSLOTS * sizeof(float);
  int*    garg  = (int*)w;     w += NROWS * sizeof(int);

  dim3 blk(256);

  // MLP: h1 = relu(x W1^T + b1), h2, h3 (h ends in buf0)
  gemm_nt_wmma<true, true><<<dim3(NROWS / 128, HDIM / 64), blk, 0, stream>>>(
      x, W1, b1, buf0, NROWS, DIN, HDIM);
  gemm_nt_wmma<true, true><<<dim3(NROWS / 128, HDIM / 64), blk, 0, stream>>>(
      buf0, W2, b2, buf1, NROWS, HDIM, HDIM);
  gemm_nt_wmma<true, true><<<dim3(NROWS / 128, HDIM / 64), blk, 0, stream>>>(
      buf1, W3, b3, buf0, NROWS, HDIM, HDIM);

  // bf16 copies of h and Kmem for the async-staged scores GEMM
  cvt_f32_bf16_kernel<<<(NROWS * HDIM / 4 + 255) / 256, blk, 0, stream>>>(
      buf0, hb, NROWS * HDIM);
  cvt_f32_bf16_kernel<<<(MSLOTS * HDIM / 4 + 255) / 256, blk, 0, stream>>>(
      Kmem, kb, MSLOTS * HDIM);

  // Scores S = h @ Kmem^T  [N, M]  (134 MB: L2-resident on 192 MB L2)
  scores_gemm_async_wmma<<<dim3(NROWS / 128, MSLOTS / 64), blk, 0, stream>>>(hb, kb, S);

  // Softmax statistics
  row_stats_kernel<<<NROWS, blk, 0, stream>>>(S, rmax, rsumP, rnmax, rsumN, garg);
  col_max_kernel<<<MSLOTS / 64, blk, 0, stream>>>(S, cmax);

  // Upload: scatter-reduce weighted queries, then normalize with memory
  zero_kernel<<<(MSLOTS * HDIM + 255) / 256, blk, 0, stream>>>(qupd, MSLOTS * HDIM);
  upload_scatter_kernel<<<NROWS, blk, 0, stream>>>(buf0, rmax, cmax, garg, qupd);
  mem_finalize_kernel<<<MSLOTS, blk, 0, stream>>>(qupd, Kmem, train, outMem);

  // Fused read/un-read: pred_min_p, pred_min_n
  attn_pv_wmma<<<dim3(NROWS / 128, HDIM / 64), blk, 0, stream>>>(
      S, Kmem, buf0, rmax, rsumP, rnmax, rsumN, outP, outN);

  // Head
  pred_dot_kernel<<<NROWS / 8, blk, 0, stream>>>(buf0, Wd, bd, outPred);
}
